// GroupedQueryAttention_79980880986810
// MI455X (gfx1250) — matrix operand order repaired and checked by reading, not compiled, not run on hardware
//
#include <hip/hip_runtime.h>
#include <hip/hip_bf16.h>

// ---------- problem constants (match reference) ----------
#define BATCH 2
#define SEQ   2048
#define DMODEL 2048
#define NHEAD 32
#define HDIM  64
#define NKVH  8
#define NQKV  3072            // 2048 (Q) + 512 (K) + 512 (V)
#define MTOT  (BATCH * SEQ)   // 4096

typedef __attribute__((ext_vector_type(16))) _Float16 v16h;
typedef __attribute__((ext_vector_type(8)))  _Float16 v8h;
typedef __attribute__((ext_vector_type(4)))  _Float16 v4h;
typedef __attribute__((ext_vector_type(8)))  float    v8f;

union frag16u { v16h v; v8h h[2]; };

// WMMA wrapper: D = A(16x32 f16) * B(32x16 f16) + C(16x16 f32)
__device__ __forceinline__ v8f wmma32(v16h a, v16h b, v8f c) {
    return __builtin_amdgcn_wmma_f32_16x16x32_f16(
        /*neg_a=*/false, a, /*neg_b=*/false, b,
        /*c_mod=*/(short)0, c, /*reuse_a=*/false, /*reuse_b=*/false);
}

// A-fragment (16x32, row-major [row][col], ld in elements).
// Lanes 0-15: row=lane, K = 0..7 then 16..23 ; lanes 16-31: K = 8..15 then 24..31.
__device__ __forceinline__ v16h load_frag_a(const _Float16* __restrict__ base,
                                            int ld, int row0, int col0, int lane) {
    const _Float16* p = base + (size_t)(row0 + (lane & 15)) * ld
                             + col0 + ((lane >> 4) << 3);
    frag16u f;
    f.h[0] = *(const v8h*)(p);
    f.h[1] = *(const v8h*)(p + 16);
    return f.v;
}

// B-fragment (32x16) from K-contiguous (transposed) storage [n][k], ld in elements.
// Lanes 0-15: col n=lane, K = 0..7 then 16..23 ; lanes 16-31: K = 8..15 then 24..31.
__device__ __forceinline__ v16h load_frag_b(const _Float16* base,
                                            int ld, int n0, int k0, int lane) {
    const _Float16* p = base + (size_t)(n0 + (lane & 15)) * ld
                             + k0 + ((lane >> 4) << 3);
    frag16u f;
    f.h[0] = *(const v8h*)(p);
    f.h[1] = *(const v8h*)(p + 16);
    return f.v;
}

// ---------- CDNA5 async global->LDS copy (ASYNCcnt tracked) ----------
__device__ __forceinline__ void async_ld_b128(void* lds_ptr, const void* gptr) {
    unsigned int loff = (unsigned int)(unsigned long long)lds_ptr;
    asm volatile("global_load_async_to_lds_b128 %0, %1, off"
                 :: "v"(loff), "v"(gptr) : "memory");
}
__device__ __forceinline__ void wait_async0() {
    asm volatile("s_wait_asynccnt 0" ::: "memory");
}

// ---------- prep kernels ----------
__global__ void cvt_x_f16(const float4* __restrict__ X, _Float16* __restrict__ Xh) {
    int i = blockIdx.x * blockDim.x + threadIdx.x;   // one float4 per thread
    float4 v = X[i];
    v4h o = { (_Float16)v.x, (_Float16)v.y, (_Float16)v.z, (_Float16)v.w };
    *(v4h*)(Xh + (size_t)i * 4) = o;
}

// Wt[n][k] = concat(Wq,Wk,Wv)[k][n], f16.  grid = (2048/256, 3072)
__global__ void build_wt(const float* __restrict__ Wq, const float* __restrict__ Wk,
                         const float* __restrict__ Wv, _Float16* __restrict__ Wt) {
    int n = blockIdx.y;
    int k = blockIdx.x * blockDim.x + threadIdx.x;
    float w;
    if (n < 2048)       w = Wq[(size_t)k * 2048 + n];
    else if (n < 2560)  w = Wk[(size_t)k * 512 + (n - 2048)];
    else                w = Wv[(size_t)k * 512 + (n - 2560)];
    Wt[(size_t)n * 2048 + k] = (_Float16)w;
}

__global__ void build_bias(const float* __restrict__ bq, const float* __restrict__ bk,
                           const float* __restrict__ bv, float* __restrict__ bias) {
    int i = blockIdx.x * blockDim.x + threadIdx.x;
    if (i >= NQKV) return;
    if (i < 2048)      bias[i] = bq[i];
    else if (i < 2560) bias[i] = bk[i - 2048];
    else               bias[i] = bv[i - 2560];
}

// ---------- fused QKV projection GEMM ----------
// C[4096,3072] = Xh[4096,2048] @ Wt^T (+bias), scattered into Q/K/Vt layouts.
// block = 256 (8 waves); wave tile = 32x64; block tile = 256x64.
// B tile (64n x 32k, 4KB) staged per block in LDS via async copy; the double-buffer
// pipeline is statically unrolled 2x so both LDS buffer addresses are constants.
// grid = (3072/64, 4096/256) = (48, 16)
__global__ void gemm_qkv(const _Float16* __restrict__ Xh,
                         const _Float16* __restrict__ Wt,
                         const float* __restrict__ bias,
                         _Float16* __restrict__ Qo,   // [B][H][S][HD]
                         _Float16* __restrict__ Ko,   // [B][KVH][S][HD]
                         _Float16* __restrict__ Vt) { // [B][KVH][HD][S]
    __shared__ alignas(16) _Float16 btile[2][64 * 32];   // 2 x 4KB

    int t    = threadIdx.x;
    int lane = t & 31;
    int wave = t >> 5;
    int n0 = blockIdx.x * 64;
    int m0 = blockIdx.y * 256 + wave * 32;

    // per-thread async-copy slot for the B tile: [n][k] f16, ld=32
    int bn = t >> 2;              // 0..63
    int bk = (t & 3) * 8;         // 0,8,16,24
    const _Float16* wsrc = Wt + (size_t)(n0 + bn) * DMODEL + bk;
    _Float16* wdst0 = &btile[0][bn * 32 + bk];
    _Float16* wdst1 = &btile[1][bn * 32 + bk];

    // prologue: fill buffer 0 with k=0..31
    async_ld_b128(wdst0, wsrc);
    wait_async0();
    __syncthreads();

    v8f acc[2][4] = {};
    auto step = [&](const _Float16* bl, int k0) {
        v16h a0 = load_frag_a(Xh, DMODEL, m0,      k0, lane);
        v16h a1 = load_frag_a(Xh, DMODEL, m0 + 16, k0, lane);
        v16h bfr[4];
#pragma unroll
        for (int j = 0; j < 4; ++j) bfr[j] = load_frag_b(bl, 32, j * 16, 0, lane);
#pragma unroll
        for (int j = 0; j < 4; ++j) {
            acc[0][j] = wmma32(a0, bfr[j], acc[0][j]);
            acc[1][j] = wmma32(a1, bfr[j], acc[1][j]);
        }
    };

    for (int k0 = 0; k0 < DMODEL; k0 += 64) {
        // stage A: prefetch k0+32 into buf1, compute k0 from buf0
        async_ld_b128(wdst1, wsrc + (k0 + 32));
        step(&btile[0][0], k0);
        wait_async0();
        __syncthreads();
        // stage B: prefetch k0+64 into buf0, compute k0+32 from buf1
        if (k0 + 64 < DMODEL)
            async_ld_b128(wdst0, wsrc + (k0 + 64));
        step(&btile[1][0], k0 + 32);
        wait_async0();
        __syncthreads();
    }

    // ---- epilogue: region & batch are block-uniform; branch-free inner stores ----
    int hi = lane >> 4;      // row offset 0/8
    int nc = lane & 15;      // column within 16-wide n tile
    int bblk  = blockIdx.y >> 3;            // batch index (256-row blocks, 2048-row batch)
    int sbase = (m0 & (SEQ - 1)) + hi * 8;  // sequence pos base for this lane

    if (n0 < 2048) {
        int hh = n0 >> 6;                    // head (n tile never straddles)
        _Float16* dq = Qo + ((size_t)(bblk * NHEAD + hh)) * SEQ * HDIM;
#pragma unroll
        for (int r = 0; r < 2; ++r)
#pragma unroll
        for (int j = 0; j < 4; ++j) {
            float bnv = bias[n0 + j * 16 + nc];
            _Float16* p = dq + (size_t)(sbase + r * 16) * HDIM + j * 16 + nc;
#pragma unroll
            for (int i = 0; i < 8; ++i)
                p[(size_t)i * HDIM] = (_Float16)(acc[r][j][i] + bnv);
        }
    } else if (n0 < 2560) {
        int kvh = (n0 - 2048) >> 6;
        _Float16* dk = Ko + ((size_t)(bblk * NKVH + kvh)) * SEQ * HDIM;
#pragma unroll
        for (int r = 0; r < 2; ++r)
#pragma unroll
        for (int j = 0; j < 4; ++j) {
            float bnv = bias[n0 + j * 16 + nc];
            _Float16* p = dk + (size_t)(sbase + r * 16) * HDIM + j * 16 + nc;
#pragma unroll
            for (int i = 0; i < 8; ++i)
                p[(size_t)i * HDIM] = (_Float16)(acc[r][j][i] + bnv);
        }
    } else {
        int kvh = (n0 - 2560) >> 6;
        _Float16* dv = Vt + ((size_t)(bblk * NKVH + kvh)) * HDIM * SEQ;
        // V is stored transposed [d][s]: the 8 per-lane values are contiguous in s
#pragma unroll
        for (int r = 0; r < 2; ++r)
#pragma unroll
        for (int j = 0; j < 4; ++j) {
            int d = j * 16 + nc;
            float bnv = bias[n0 + d];
            v8h pk;
#pragma unroll
            for (int i = 0; i < 8; ++i) pk[i] = (_Float16)(acc[r][j][i] + bnv);
            *(v8h*)(dv + (size_t)d * SEQ + sbase + r * 16) = pk;   // global_store_b128
        }
    }
}

// ---------- flash attention ----------
// block = 256 (8 waves, all same (b,h)); wave handles 16 queries -> block = 128.
// K/V key-chunks (32 keys, 4KB each) staged per block in LDS via async copy,
// statically 2x-unrolled double buffering; shared by all 8 waves.
// grid = B*H*(S/128) = 1024
__global__ void attn_kernel(const _Float16* __restrict__ Q,
                            const _Float16* __restrict__ Kd,
                            const _Float16* __restrict__ Vt,
                            float* __restrict__ out) {
    __shared__ alignas(16) _Float16 ktile[2][32 * 64];   // [key][d]  2 x 4KB
    __shared__ alignas(16) _Float16 vtile[2][64 * 32];   // [d][key]  2 x 4KB
    __shared__ alignas(16) _Float16 pbuf[8][16 * 32];    // per-wave P staging, 8KB

    int t    = threadIdx.x;
    int lane = t & 31;
    int wave = t >> 5;
    int bid  = blockIdx.x;
    int qt   = bid & 15;          // S/128 q-tiles
    int bh   = bid >> 4;          // 0..63
    int b    = bh >> 5;
    int h    = bh & 31;
    int kv   = h >> 2;            // h / REP
    int q0   = qt * 128 + wave * 16;

    const _Float16* Qbase = Q  + ((size_t)(b * NHEAD + h)) * SEQ * HDIM;   // [s][d]
    const _Float16* Kbase = Kd + ((size_t)(b * NKVH + kv)) * SEQ * HDIM;   // [s][d]
    const _Float16* Vbase = Vt + ((size_t)(b * NKVH + kv)) * HDIM * SEQ;   // [d][s]

    // per-thread async-copy slots
    int kkey = t >> 3;            // 0..31
    int kdc  = (t & 7) * 8;       // d chunk
    const _Float16* ksrc = Kbase + (size_t)kkey * HDIM + kdc;
    _Float16* kdst0 = &ktile[0][kkey * 64 + kdc];
    _Float16* kdst1 = &ktile[1][kkey * 64 + kdc];

    int vd = t >> 2;              // 0..63
    int vkc = (t & 3) * 8;        // key chunk
    const _Float16* vsrc = Vbase + (size_t)vd * SEQ + vkc;
    _Float16* vdst0 = &vtile[0][vd * 32 + vkc];
    _Float16* vdst1 = &vtile[1][vd * 32 + vkc];

    // Q A-fragments for d=0..31 and d=32..63 (kept resident)
    v16h qa0 = load_frag_a(Qbase, HDIM, q0, 0, lane);
    v16h qa1 = load_frag_a(Qbase, HDIM, q0, 32, lane);

    // prologue: fill buffer 0 with keys 0..31
    async_ld_b128(kdst0, ksrc);
    async_ld_b128(vdst0, vsrc);
    wait_async0();
    __syncthreads();

    v8f ctx[4] = {};
    float mrun[8], lrun[8], corr[8];
#pragma unroll
    for (int i = 0; i < 8; ++i) { mrun[i] = -1e30f; lrun[i] = 0.0f; }

    int hi = lane >> 4;
    int nc = lane & 15;

    auto step = [&](const _Float16* kl, const _Float16* vl) {
        // ---- scores for 32 keys: preload all K frags, then 4 WMMAs ----
        v16h kf[4];
        kf[0] = load_frag_b(kl, 64, 0,  0,  lane);
        kf[1] = load_frag_b(kl, 64, 0,  32, lane);
        kf[2] = load_frag_b(kl, 64, 16, 0,  lane);
        kf[3] = load_frag_b(kl, 64, 16, 32, lane);
        v8f s0 = {}, s1 = {};
        s0 = wmma32(qa0, kf[0], s0);
        s0 = wmma32(qa1, kf[1], s0);
        s1 = wmma32(qa0, kf[2], s1);
        s1 = wmma32(qa1, kf[3], s1);

        // ---- online softmax (rows live across 16-lane halves) ----
#pragma unroll
        for (int i = 0; i < 8; ++i) {
            float a = s0[i] * 0.125f;          // 1/sqrt(64)
            float c = s1[i] * 0.125f;
            float mx = fmaxf(a, c);
#pragma unroll
            for (int off = 1; off < 16; off <<= 1)
                mx = fmaxf(mx, __shfl_xor(mx, off, 32));
            float mnew = fmaxf(mrun[i], mx);
            float p0 = __expf(a - mnew);
            float p1 = __expf(c - mnew);
            float rs = p0 + p1;
#pragma unroll
            for (int off = 1; off < 16; off <<= 1)
                rs += __shfl_xor(rs, off, 32);
            float cr = __expf(mrun[i] - mnew);
            corr[i] = cr;
            lrun[i] = lrun[i] * cr + rs;
            mrun[i] = mnew;
            s0[i] = p0; s1[i] = p1;            // reuse as P
        }

        // ---- C-layout -> A-layout transpose of P via per-wave LDS ----
        _Float16* pw = &pbuf[wave][0];
#pragma unroll
        for (int i = 0; i < 8; ++i) {
            int r = i + hi * 8;
            pw[r * 32 + nc]      = (_Float16)s0[i];
            pw[r * 32 + 16 + nc] = (_Float16)s1[i];
        }
        asm volatile("s_wait_dscnt 0" ::: "memory");

        frag16u pf;
        {
            int r  = lane & 15;
            int c0 = hi * 8;
            pf.h[0] = *(const v8h*)&pbuf[wave][r * 32 + c0];
            pf.h[1] = *(const v8h*)&pbuf[wave][r * 32 + c0 + 16];
        }

        // ---- ctx = ctx*corr + P @ V (preload all V frags, then 4 WMMAs) ----
#pragma unroll
        for (int j = 0; j < 4; ++j)
#pragma unroll
            for (int i = 0; i < 8; ++i) ctx[j][i] *= corr[i];
        v16h vf[4];
#pragma unroll
        for (int j = 0; j < 4; ++j) vf[j] = load_frag_b(vl, 32, j * 16, 0, lane);
#pragma unroll
        for (int j = 0; j < 4; ++j) ctx[j] = wmma32(pf.v, vf[j], ctx[j]);
    };

    for (int k0 = 0; k0 < SEQ; k0 += 64) {
        // stage A: prefetch keys k0+32 into buf1, compute keys k0 from buf0
        async_ld_b128(kdst1, ksrc + (size_t)(k0 + 32) * HDIM);
        async_ld_b128(vdst1, vsrc + (k0 + 32));
        step(&ktile[0][0], &vtile[0][0]);
        wait_async0();
        __syncthreads();
        // stage B: prefetch keys k0+64 into buf0, compute keys k0+32 from buf1
        if (k0 + 64 < SEQ) {
            async_ld_b128(kdst0, ksrc + (size_t)(k0 + 64) * HDIM);
            async_ld_b128(vdst0, vsrc + (k0 + 64));
        }
        step(&ktile[1][0], &vtile[1][0]);
        wait_async0();
        __syncthreads();
    }

    // ---- normalize and store fp32 output [b][s][h*64+d] ----
    float inv[8];
#pragma unroll
    for (int i = 0; i < 8; ++i) inv[i] = 1.0f / lrun[i];
    float* obase = out + (size_t)b * SEQ * DMODEL + (size_t)h * HDIM;
#pragma unroll
    for (int j = 0; j < 4; ++j) {
#pragma unroll
        for (int i = 0; i < 8; ++i) {
            int s = q0 + i + hi * 8;
            obase[(size_t)s * DMODEL + j * 16 + nc] = ctx[j][i] * inv[i];
        }
    }
}

// ---------- host launcher ----------
extern "C" void kernel_launch(void* const* d_in, const int* in_sizes, int n_in,
                              void* d_out, int out_size, void* d_ws, size_t ws_size,
                              hipStream_t stream) {
    const float* X  = (const float*)d_in[0];
    const float* Wq = (const float*)d_in[1];
    const float* bq = (const float*)d_in[2];
    const float* Wk = (const float*)d_in[3];
    const float* bk = (const float*)d_in[4];
    const float* Wv = (const float*)d_in[5];
    const float* bv = (const float*)d_in[6];
    float* out = (float*)d_out;

    char* ws = (char*)d_ws;
    size_t off = 0;
    _Float16* Xh = (_Float16*)(ws + off); off += (size_t)MTOT * DMODEL * 2;        // 16 MB
    _Float16* Wt = (_Float16*)(ws + off); off += (size_t)NQKV * DMODEL * 2;        // 12 MB
    float* bias  = (float*)(ws + off);    off += (size_t)NQKV * 4;                 // 12 KB
    _Float16* Qb = (_Float16*)(ws + off); off += (size_t)BATCH * NHEAD * SEQ * HDIM * 2; // 16 MB
    _Float16* Kb = (_Float16*)(ws + off); off += (size_t)BATCH * NKVH * SEQ * HDIM * 2;  // 4 MB
    _Float16* Vb = (_Float16*)(ws + off); off += (size_t)BATCH * NKVH * HDIM * SEQ * 2;  // 4 MB
    (void)ws_size; (void)in_sizes; (void)n_in; (void)out_size;

    // 1) downconvert activations
    cvt_x_f16<<<(MTOT * DMODEL) / (256 * 4), 256, 0, stream>>>((const float4*)X, Xh);
    // 2) fuse + transpose weights
    build_wt<<<dim3(DMODEL / 256, NQKV), 256, 0, stream>>>(Wq, Wk, Wv, Wt);
    // 3) fuse biases
    build_bias<<<(NQKV + 255) / 256, 256, 0, stream>>>(bq, bk, bv, bias);
    // 4) fused QKV projection (WMMA + async LDS staging)
    gemm_qkv<<<dim3(NQKV / 64, MTOT / 256), 256, 0, stream>>>(Xh, Wt, bias, Qb, Kb, Vb);
    // 5) flash attention (WMMA + async LDS staging)
    attn_kernel<<<BATCH * NHEAD * (SEQ / 128), 256, 0, stream>>>(Qb, Kb, Vb, out);
}
